// FrameQuerySwin_87316685128169
// MI455X (gfx1250) — matrix operand order repaired and checked by reading, not compiled, not run on hardware
//
#include <hip/hip_runtime.h>
#include <cstddef>

// ---------------------------------------------------------------------------
// Types for CDNA5 WMMA (wave32): v_wmma_f32_16x16x32_bf16
// ---------------------------------------------------------------------------
typedef __attribute__((ext_vector_type(16))) __bf16 v16bf;
typedef __attribute__((ext_vector_type(8)))  float  v8f;

union Frag16 { v16bf v; unsigned u[8]; unsigned short e[16]; };

// Pack two fp32 into one dword of two bf16 (RNE): 2 adds + 1 v_perm_b32.
__device__ __forceinline__ unsigned pack2bf(float lo, float hi) {
    union { float f; unsigned u; } a, b;
    a.f = lo; b.f = hi;
    unsigned ua = a.u + (0x7fffu + ((a.u >> 16) & 1u));
    unsigned ub = b.u + (0x7fffu + ((b.u >> 16) & 1u));
    return __builtin_amdgcn_perm(ub, ua, 0x07060302);  // {hi16(ub), hi16(ua)}
}

__device__ __forceinline__ unsigned short f2bfu(float x) {
    union { float f; unsigned u; } a; a.f = x;
    return (unsigned short)((a.u + (0x7fffu + ((a.u >> 16) & 1u))) >> 16);
}

__device__ __forceinline__ void pack8(Frag16& f, int base, float4 x, float4 y) {
    f.u[base + 0] = pack2bf(x.x, x.y);
    f.u[base + 1] = pack2bf(x.z, x.w);
    f.u[base + 2] = pack2bf(y.x, y.y);
    f.u[base + 3] = pack2bf(y.z, y.w);
}

// A-fragment (16x32) from f32 (used for softmax P in LDS only)
__device__ __forceinline__ v16bf load_afrag_f32(const float* row, int kb, int grp) {
    const float4* p = reinterpret_cast<const float4*>(row + kb + 16 * grp);
    const float4* q = reinterpret_cast<const float4*>(row + kb + 16 * grp + 8);
    Frag16 f;
    pack8(f, 0, p[0], p[1]);
    pack8(f, 4, q[0], q[1]);
    return f.v;
}

// A-fragment (16x32) from bf16: two 16-byte loads, zero ALU.
__device__ __forceinline__ v16bf load_afrag_bf16(const unsigned short* row, int kb, int grp) {
    const uint4* p = reinterpret_cast<const uint4*>(row + kb + 16 * grp);      // K [16g, +8)
    const uint4* q = reinterpret_cast<const uint4*>(row + kb + 16 * grp + 8);  // K [16g+8, +8)
    uint4 a = p[0], b = q[0];
    Frag16 f;
    f.u[0] = a.x; f.u[1] = a.y; f.u[2] = a.z; f.u[3] = a.w;
    f.u[4] = b.x; f.u[5] = b.y; f.u[6] = b.z; f.u[7] = b.w;
    return f.v;
}

// B-fragment (32x16) from bf16: 16 contiguous K values = 2x 16-byte loads.
__device__ __forceinline__ v16bf load_bfrag_bf16(const unsigned short* __restrict__ colbase, int kb) {
    const uint4* p = reinterpret_cast<const uint4*>(colbase + kb);
    uint4 a = p[0], b = p[1];
    Frag16 f;
    f.u[0] = a.x; f.u[1] = a.y; f.u[2] = a.z; f.u[3] = a.w;
    f.u[4] = b.x; f.u[5] = b.y; f.u[6] = b.z; f.u[7] = b.w;
    return f.v;
}

__device__ __forceinline__ v8f wmma_bf16(v16bf a, v16bf b, v8f c) {
    return __builtin_amdgcn_wmma_f32_16x16x32_bf16(false, a, false, b, (short)0, c, false, false);
}

// ---------------------------------------------------------------------------
// fp32 -> bf16 conversion (weights): thread packs 8 floats -> uint4. Exact grids.
// ---------------------------------------------------------------------------
__global__ void cvt_bf16_kernel(const float4* __restrict__ src, uint4* __restrict__ dst)
{
    const int i = blockIdx.x * 256 + threadIdx.x;
    float4 x = src[2 * i], y = src[2 * i + 1];
    uint4 o;
    o.x = pack2bf(x.x, x.y);
    o.y = pack2bf(x.z, x.w);
    o.z = pack2bf(y.x, y.y);
    o.w = pack2bf(y.z, y.w);
    dst[i] = o;
}

// ---------------------------------------------------------------------------
// GEMM: out = A[M,K] @ W[N,K]^T + bias (+ReLU). A,W bf16; out f32 and/or bf16.
// Wave tile 32(M) x 64(N): 8 accumulators, 8 WMMAs per K-step, all operands
// pure b128 loads. grid = (N/256, M/32), block = 128.
// ---------------------------------------------------------------------------
template<int RELU, int STOREF, int STOREBF>
__global__ __launch_bounds__(128) void gemm_wmma_kernel(
    const unsigned short* __restrict__ A, const unsigned short* __restrict__ W,
    const float* __restrict__ bias, float* __restrict__ outf,
    unsigned short* __restrict__ outbf, int M, int N, int K)
{
    const int wave = threadIdx.x >> 5;
    const int lane = threadIdx.x & 31;
    const int grp  = lane >> 4;
    const int ln   = lane & 15;
    const int m0   = blockIdx.y * 32;
    const int n0   = (blockIdx.x * 4 + wave) * 64;

    const unsigned short* a0 = A + (size_t)(m0 + ln) * K;
    const unsigned short* a1 = a0 + (size_t)16 * K;
    const unsigned short* w0 = W + (size_t)(n0 + ln) * K + 16 * grp;
    const unsigned short* w1 = w0 + (size_t)16 * K;
    const unsigned short* w2 = w1 + (size_t)16 * K;
    const unsigned short* w3 = w2 + (size_t)16 * K;

    v8f c00 = {}, c01 = {}, c02 = {}, c03 = {};
    v8f c10 = {}, c11 = {}, c12 = {}, c13 = {};

#pragma unroll 4
    for (int kb = 0; kb < K; kb += 32) {
        v16bf af0 = load_afrag_bf16(a0, kb, grp);
        v16bf af1 = load_afrag_bf16(a1, kb, grp);
        v16bf b0 = load_bfrag_bf16(w0, kb);
        v16bf b1 = load_bfrag_bf16(w1, kb);
        v16bf b2 = load_bfrag_bf16(w2, kb);
        v16bf b3 = load_bfrag_bf16(w3, kb);
        c00 = wmma_bf16(af0, b0, c00);
        c10 = wmma_bf16(af1, b0, c10);
        c01 = wmma_bf16(af0, b1, c01);
        c11 = wmma_bf16(af1, b1, c11);
        c02 = wmma_bf16(af0, b2, c02);
        c12 = wmma_bf16(af1, b2, c12);
        c03 = wmma_bf16(af0, b3, c03);
        c13 = wmma_bf16(af1, b3, c13);
    }

    const int mr0 = m0 + 8 * grp;
#pragma unroll
    for (int j = 0; j < 4; ++j) {
        const int ncol = n0 + j * 16 + ln;
        const float bv = bias[ncol];
        v8f accA = (j == 0) ? c00 : (j == 1) ? c01 : (j == 2) ? c02 : c03;
        v8f accB = (j == 0) ? c10 : (j == 1) ? c11 : (j == 2) ? c12 : c13;
#pragma unroll
        for (int r = 0; r < 8; ++r) {
            float v = accA[r] + bv;
            if (RELU) v = fmaxf(v, 0.0f);
            if (STOREF)  outf[(size_t)(mr0 + r) * N + ncol] = v;
            if (STOREBF) outbf[(size_t)(mr0 + r) * N + ncol] = f2bfu(v);
        }
#pragma unroll
        for (int r = 0; r < 8; ++r) {
            float v = accB[r] + bv;
            if (RELU) v = fmaxf(v, 0.0f);
            if (STOREF)  outf[(size_t)(mr0 + 16 + r) * N + ncol] = v;
            if (STOREBF) outbf[(size_t)(mr0 + 16 + r) * N + ncol] = f2bfu(v);
        }
    }
}

// ---------------------------------------------------------------------------
// Attention: one block per (b, h, 16-query tile). QKV is bf16 [NT,768].
// MODE 0: key-padding mask (key time >= 37 -> score = -1e9)
// MODE 1: shifted-window additive mask (-1000 on blocked (wq,wk), nw==0)
// Output: f32 (for residual path) + bf16 shadow (proj GEMM input).
// ---------------------------------------------------------------------------
template<int MODE>
__global__ __launch_bounds__(128) void attn_kernel(
    const unsigned short* __restrict__ qkv, float* __restrict__ o,
    unsigned short* __restrict__ obf)
{
    __shared__ float sS[16][512];      // score / prob rows (32 KB)
    __shared__ float sRed[16][8];
    __shared__ float sRow[16];
    __shared__ float sP[2][8][32];     // partial O reduction

    const int S = 512, LD = 768;
    const int b  = blockIdx.z;
    const int h  = blockIdx.y;
    const int q0 = blockIdx.x * 16;
    const int tid  = threadIdx.x;
    const int wave = tid >> 5;
    const int lane = tid & 31;
    const int grp  = lane >> 4;
    const int ln   = lane & 15;
    const int nw   = b % 5;
    const int wq   = q0 >> 6;          // tile-uniform query window row
    const float scale = 0.17677669529663687f;  // 1/sqrt(32)

    // ---- Q fragment: 2x b128 loads ----
    const unsigned short* qrow = qkv + (size_t)(b * S + q0 + ln) * LD + h * 32;
    v16bf qf = load_afrag_bf16(qrow, 0, grp);

    // ---- Phase 1: 32 key tiles of scores split across 4 waves ----
#pragma unroll
    for (int i = 0; i < 8; ++i) {
        const int kt = wave + 4 * i;
        const int sidx = kt * 16 + ln;             // key column for this lane
        const unsigned short* kcol =
            qkv + (size_t)(b * S + kt * 16 + ln) * LD + 256 + h * 32 + 16 * grp;
        v16bf kf = load_bfrag_bf16(kcol, 0);
        v8f acc = {};
        acc = wmma_bf16(qf, kf, acc);

        bool  kill = false;
        float madd = 0.0f;
        if (MODE == 0) {
            kill = (nw * 8 + (sidx >> 6)) >= 37;
        } else {
            if (nw == 0) {
                const int wk = sidx >> 6;
                const bool aq = (wq >= 1 && wq <= 3);
                const bool ak = (wk >= 1 && wk <= 3);
                const bool mk = aq || ak || (wq < 4 && wk >= 4) || (wq >= 4 && wk < 4);
                madd = mk ? -1000.0f : 0.0f;
            }
        }
#pragma unroll
        for (int r = 0; r < 8; ++r) {
            float v = acc[r] * scale;
            if (MODE == 0) v = kill ? -1e9f : v;
            else           v += madd;
            sS[r + 8 * grp][sidx] = v;
        }
    }
    __syncthreads();

    // ---- Phase 2: softmax over 512-wide rows, float4 vectorized ----
    const int row = tid >> 3, sub = tid & 7;
    float4* rowp = reinterpret_cast<float4*>(&sS[row][0]);
    float mx = -3.4e38f;
    for (int i = sub; i < 128; i += 8) {
        float4 x = rowp[i];
        mx = fmaxf(mx, fmaxf(fmaxf(x.x, x.y), fmaxf(x.z, x.w)));
    }
    sRed[row][sub] = mx;
    __syncthreads();
    if (tid < 16) {
        float m2 = sRed[tid][0];
        for (int i = 1; i < 8; ++i) m2 = fmaxf(m2, sRed[tid][i]);
        sRow[tid] = m2;
    }
    __syncthreads();
    const float rm = sRow[row];
    float sm = 0.f;
    for (int i = sub; i < 128; i += 8) {
        float4 x = rowp[i];
        x.x = __expf(x.x - rm); x.y = __expf(x.y - rm);
        x.z = __expf(x.z - rm); x.w = __expf(x.w - rm);
        rowp[i] = x;
        sm += (x.x + x.y) + (x.z + x.w);
    }
    sRed[row][sub] = sm;
    __syncthreads();
    if (tid < 16) {
        float s2 = 0.f;
        for (int i = 0; i < 8; ++i) s2 += sRed[tid][i];
        sRow[tid] = s2;
    }
    __syncthreads();
    const float inv = 1.0f / sRow[row];
    for (int i = sub; i < 128; i += 8) {
        float4 x = rowp[i];
        x.x *= inv; x.y *= inv; x.z *= inv; x.w *= inv;
        rowp[i] = x;
    }
    __syncthreads();

    // ---- Phase 3: O = P @ V. wave -> (dtile = w&1, K-half = w>>1) ----
    const int dtile = wave & 1, khalf = wave >> 1;
    v8f oacc = {};
#pragma unroll
    for (int i = 0; i < 8; ++i) {
        const int kt2 = khalf * 8 + i;
        v16bf pa = load_afrag_f32(&sS[ln][0], kt2 * 32, grp);
        const unsigned short* vbase =
            qkv + (size_t)(b * S + kt2 * 32 + 16 * grp) * LD + 512 + h * 32 + dtile * 16 + ln;
        Frag16 vf;
#pragma unroll
        for (int t = 0; t < 16; ++t)
            vf.e[t] = vbase[(size_t)t * LD];
        oacc = wmma_bf16(pa, vf.v, oacc);
    }
    if (khalf == 1) {
#pragma unroll
        for (int r = 0; r < 8; ++r) sP[dtile][r][lane] = oacc[r];
    }
    __syncthreads();
    if (khalf == 0) {
#pragma unroll
        for (int r = 0; r < 8; ++r) {
            float val = oacc[r] + sP[dtile][r][lane];
            const size_t oi = (size_t)(b * S + q0 + r + 8 * grp) * 256 + h * 32 + dtile * 16 + ln;
            o[oi]   = val;
            obf[oi] = f2bfu(val);
        }
    }
}

// ---------------------------------------------------------------------------
// Residual + LayerNorm over C=256: out = LN(x + a) * g + b (+bf16 shadow)
// ---------------------------------------------------------------------------
template<int SHADOW>
__global__ __launch_bounds__(256) void add_ln_kernel(
    const float* __restrict__ x, const float* __restrict__ a,
    const float* __restrict__ g, const float* __restrict__ bb,
    float* __restrict__ out, unsigned short* __restrict__ outbf)
{
    __shared__ float red[256];
    const int tid = threadIdx.x;
    const size_t idx = (size_t)blockIdx.x * 256 + tid;
    const float s = x[idx] + a[idx];
    red[tid] = s;
    __syncthreads();
    for (int off = 128; off > 0; off >>= 1) {
        if (tid < off) red[tid] += red[tid + off];
        __syncthreads();
    }
    const float mu = red[0] * (1.0f / 256.0f);
    __syncthreads();
    const float d = s - mu;
    red[tid] = d * d;
    __syncthreads();
    for (int off = 128; off > 0; off >>= 1) {
        if (tid < off) red[tid] += red[tid + off];
        __syncthreads();
    }
    const float var = red[0] * (1.0f / 256.0f);
    const float v = d * rsqrtf(var + 1e-5f) * g[tid] + bb[tid];
    out[idx] = v;
    if (SHADOW) outbf[idx] = f2bfu(v);
}

// ---------------------------------------------------------------------------
// float4 gathers (f32 + bf16 shadow where the result feeds a GEMM A operand)
// ---------------------------------------------------------------------------
__global__ void gather_win0_kernel(const float4* __restrict__ fq,
                                   float4* __restrict__ xw, uint2* __restrict__ xwbf)
{
    const int idx = blockIdx.x * 256 + threadIdx.x;   // float4 index
    const int c4 = idx & 63;
    const int s  = (idx >> 6) & 511;
    const int bi = idx >> 15;
    const int lb = bi / 5, nw = bi % 5;
    const int w = s >> 6, f = s & 63;
    const int t = nw * 8 + w;
    float4 v = make_float4(0.f, 0.f, 0.f, 0.f);
    if (t < 37) v = fq[(((size_t)t * 64 + f) * 8 + lb) * 64 + c4];
    xw[idx] = v;
    uint2 sb; sb.x = pack2bf(v.x, v.y); sb.y = pack2bf(v.z, v.w);
    xwbf[idx] = sb;
}

__global__ void regather_kernel(const float4* __restrict__ src,
                                float4* __restrict__ dst, uint2* __restrict__ dstbf)
{
    const int idx = blockIdx.x * 256 + threadIdx.x;
    const int c4 = idx & 63;
    const int s  = (idx >> 6) & 511;
    const int bi = idx >> 15;
    const int lb = bi / 5, nw1 = bi % 5;
    const int w1 = s >> 6, f = s & 63;
    const int t  = (nw1 * 8 + w1 + 36) % 40;
    const int nw0 = t >> 3, w0 = t & 7;
    float4 v = src[((size_t)(lb * 5 + nw0) * 512 + w0 * 64 + f) * 64 + c4];
    dst[idx] = v;
    uint2 sb; sb.x = pack2bf(v.x, v.y); sb.y = pack2bf(v.z, v.w);
    dstbf[idx] = sb;
}

__global__ void final_gather_kernel(const float4* __restrict__ src, float4* __restrict__ out)
{
    const int idx = blockIdx.x * 256 + threadIdx.x;
    const int c4 = idx & 63;
    const int lb = (idx >> 6) & 7;
    const int tf = idx >> 9;
    const int f = tf & 63, t = tf >> 6;          // t < 37
    const int tr = (t + 4) % 40;
    const int nw1 = tr >> 3, w1 = tr & 7;
    out[idx] = src[((size_t)(lb * 5 + nw1) * 512 + w1 * 64 + f) * 64 + c4];
}

// ---------------------------------------------------------------------------
// Host orchestration
// ---------------------------------------------------------------------------
extern "C" void kernel_launch(void* const* d_in, const int* in_sizes, int n_in,
                              void* d_out, int out_size, void* d_ws, size_t ws_size,
                              hipStream_t stream)
{
    (void)in_sizes; (void)n_in; (void)out_size; (void)ws_size;
    const float* fq   = (const float*)d_in[0];
    const float* qkvw = (const float*)d_in[1];   // [2, 768, 256]
    const float* qkvb = (const float*)d_in[2];   // [2, 768]
    const float* outw = (const float*)d_in[3];   // [2, 256, 256]
    const float* outb = (const float*)d_in[4];   // [2, 256]
    const float* ln1g = (const float*)d_in[5];
    const float* ln1b = (const float*)d_in[6];
    const float* w1   = (const float*)d_in[7];   // [2, 2048, 256]
    const float* b1   = (const float*)d_in[8];   // [2, 2048]
    const float* w2   = (const float*)d_in[9];   // [2, 256, 2048]
    const float* b2   = (const float*)d_in[10];  // [2, 256]
    const float* ln2g = (const float*)d_in[11];
    const float* ln2b = (const float*)d_in[12];

    const size_t NT = 20480;                     // 40 batches * 512 tokens
    float* ws    = (float*)d_ws;
    // fp32 buffers (residual / LN path)
    float* bufX  = ws;                           // activation, window layout
    float* bufY  = bufX  + NT * 256;             // post-attention LN
    float* bufP  = bufY  + NT * 256;             // proj / FFN2 result
    float* bufX2 = bufP  + NT * 256;             // post-FFN LN
    float* bufO  = bufX2 + NT * 256;             // attention output

    // bf16 buffers (all GEMM/attention matrix operands)
    unsigned short* bf = (unsigned short*)(bufO + NT * 256);
    unsigned short* bufX_bf  = bf;               // NT*256
    unsigned short* bufY_bf  = bufX_bf + NT * 256;
    unsigned short* bufO_bf  = bufY_bf + NT * 256;
    unsigned short* bufQKV_bf = bufO_bf + NT * 256;    // NT*768 (bf16 only)
    unsigned short* bufH_bf   = bufQKV_bf + NT * 768;  // NT*2048 (bf16 only)
    // bf16 weights
    unsigned short* qkvw_bf = bufH_bf + NT * 2048;             // 2*768*256
    unsigned short* outw_bf = qkvw_bf + (size_t)2 * 768 * 256; // 2*256*256
    unsigned short* w1_bf   = outw_bf + (size_t)2 * 256 * 256; // 2*2048*256
    unsigned short* w2_bf   = w1_bf   + (size_t)2 * 2048 * 256;// 2*256*2048

    // pre-convert weights to bf16 (8 floats / thread; exact grids)
    cvt_bf16_kernel<<<192, 256, 0, stream>>>((const float4*)qkvw, (uint4*)qkvw_bf);
    cvt_bf16_kernel<<<64,  256, 0, stream>>>((const float4*)outw, (uint4*)outw_bf);
    cvt_bf16_kernel<<<512, 256, 0, stream>>>((const float4*)w1,   (uint4*)w1_bf);
    cvt_bf16_kernel<<<512, 256, 0, stream>>>((const float4*)w2,   (uint4*)w2_bf);

    gather_win0_kernel<<<5120, 256, 0, stream>>>((const float4*)fq, (float4*)bufX, (uint2*)bufX_bf);

    for (int l = 0; l < 2; ++l) {
        // QKV projection -> bf16 only (attention is the sole consumer)
        gemm_wmma_kernel<0, 0, 1><<<dim3(3, 640), 128, 0, stream>>>(
            bufX_bf, qkvw_bf + (size_t)l * 768 * 256, qkvb + l * 768,
            nullptr, bufQKV_bf, (int)NT, 768, 256);
        // windowed multi-head attention -> f32 + bf16 shadow
        if ((l & 1) == 0)
            attn_kernel<0><<<dim3(32, 8, 40), 128, 0, stream>>>(bufQKV_bf, bufO, bufO_bf);
        else
            attn_kernel<1><<<dim3(32, 8, 40), 128, 0, stream>>>(bufQKV_bf, bufO, bufO_bf);
        // output projection -> f32 (residual input)
        gemm_wmma_kernel<0, 1, 0><<<dim3(1, 640), 128, 0, stream>>>(
            bufO_bf, outw_bf + (size_t)l * 256 * 256, outb + l * 256,
            bufP, nullptr, (int)NT, 256, 256);
        // residual + LN1 -> f32 + bf16 shadow (FFN1 input)
        add_ln_kernel<1><<<NT, 256, 0, stream>>>(bufX, bufP, ln1g + l * 256, ln1b + l * 256,
                                                 bufY, bufY_bf);
        // FFN1 -> bf16 only (FFN2 is the sole consumer)
        gemm_wmma_kernel<1, 0, 1><<<dim3(8, 640), 128, 0, stream>>>(
            bufY_bf, w1_bf + (size_t)l * 2048 * 256, b1 + l * 2048,
            nullptr, bufH_bf, (int)NT, 2048, 256);
        // FFN2 -> f32 (residual input)
        gemm_wmma_kernel<0, 1, 0><<<dim3(1, 640), 128, 0, stream>>>(
            bufH_bf, w2_bf + (size_t)l * 256 * 2048, b2 + l * 256,
            bufP, nullptr, (int)NT, 256, 2048);
        // residual + LN2 -> f32
        add_ln_kernel<0><<<NT, 256, 0, stream>>>(bufY, bufP, ln2g + l * 256, ln2b + l * 256,
                                                 bufX2, nullptr);

        if (l == 0)
            regather_kernel<<<5120, 256, 0, stream>>>((const float4*)bufX2,
                                                      (float4*)bufX, (uint2*)bufX_bf);
        else
            final_gather_kernel<<<4736, 256, 0, stream>>>((const float4*)bufX2, (float4*)d_out);
    }
}
